// Residual_GCN_7069516169710
// MI455X (gfx1250) — matrix operand order repaired and checked by reading, not compiled, not run on hardware
//
#include <hip/hip_runtime.h>
#include <hip/hip_bf16.h>
#include <cstdint>
#include <cstddef>

typedef __attribute__((ext_vector_type(16))) _Float16 v16h;
typedef __attribute__((ext_vector_type(8)))  _Float16 v8h;
typedef __attribute__((ext_vector_type(8)))  float    v8f;

#define NEG_SLOPE 0.2f

// ---------------------------------------------------------------- conversions
__global__ void k_cvt_f16(const float* __restrict__ in, _Float16* __restrict__ out, int n) {
  int i = blockIdx.x * blockDim.x + threadIdx.x;
  if (i < n) out[i] = (_Float16)in[i];
}

// W: [K x Nc] row-major f32  ->  Wt: [Nc x K] f16 (so B-operand loads are contiguous)
__global__ void k_cvt_transpose_f16(const float* __restrict__ W, _Float16* __restrict__ Wt,
                                    int K, int Nc) {
  int i = blockIdx.x * blockDim.x + threadIdx.x;
  if (i < K * Nc) {
    int k = i / Nc, n = i % Nc;
    Wt[(size_t)n * K + k] = (_Float16)W[i];
  }
}

__global__ void k_fill_u32(unsigned* __restrict__ p, unsigned v, int n) {
  int i = blockIdx.x * blockDim.x + threadIdx.x;
  if (i < n) p[i] = v;
}

// ---------------------------------------------------------------- WMMA GEMM
// C[M x Nc] (f32) = A[M x K] (f16 row-major) x B[K x Nc], with B given transposed
// as Bt[Nc x K].  One wave per 16x16 C tile; K-loop of v_wmma_f32_16x16x32_f16.
//
// A frag (16-bit A 16x32, ISA 7.12.2): lane l holds row M=l&15;
//   elements 0..7  = K[(l>>4)*8      .. +7]
//   elements 8..15 = K[16+(l>>4)*8   .. +7]
// B frag (16-bit B 32x16): lane l holds col N=l&15; same K order as A frag
// C/D (f32 16x16): VGPR v, lane l -> row M = v + 8*(l>>4), col N = l&15
__global__ void k_wmma_gemm(const _Float16* __restrict__ A, const _Float16* __restrict__ Bt,
                            float* __restrict__ C, int M, int K, int Nc) {
  const int lane = threadIdx.x & 31;
  const int wave = threadIdx.x >> 5;
  const int ntn  = Nc >> 4;
  const int tile = blockIdx.x * (blockDim.x >> 5) + wave;
  if (tile >= (M >> 4) * ntn) return;          // wave-uniform guard (EXEC stays all-1)
  const int mbase = (tile / ntn) << 4;
  const int nbase = (tile % ntn) << 4;
  const int l15 = lane & 15;
  const int hi  = lane >> 4;

  const _Float16* ap = A  + (size_t)(mbase + l15) * K + hi * 8;
  const _Float16* bp = Bt + (size_t)(nbase + l15) * K + hi * 8;

  v8f acc = {};
  for (int kb = 0; kb < K; kb += 32) {
    v8h a0 = *(const v8h*)(ap + kb);
    v8h a1 = *(const v8h*)(ap + kb + 16);
    v16h af = __builtin_shufflevector(a0, a1, 0,1,2,3,4,5,6,7,8,9,10,11,12,13,14,15);
    v8h b0 = *(const v8h*)(bp + kb);
    v8h b1 = *(const v8h*)(bp + kb + 16);
    v16h bf = __builtin_shufflevector(b0, b1, 0,1,2,3,4,5,6,7,8,9,10,11,12,13,14,15);
    acc = __builtin_amdgcn_wmma_f32_16x16x32_f16(false, af, false, bf,
                                                 (short)0, acc, false, false);
  }
  float* cp = C + (size_t)(mbase + hi * 8) * Nc + nbase + l15;
#pragma unroll
  for (int v = 0; v < 8; ++v) cp[(size_t)v * Nc] = acc[v];
}

// ---------------------------------------------------------------- attention
__global__ void k_alpha(const float* __restrict__ h, const float* __restrict__ avs,
                        const float* __restrict__ avd, float* __restrict__ as_,
                        float* __restrict__ ad_, int n, int H) {
  int i = blockIdx.x * blockDim.x + threadIdx.x;
  if (i >= n) return;
  float s = 0.f, d = 0.f;
  const float* hp = h + (size_t)i * H;
  for (int k = 0; k < H; ++k) { float v = hp[k]; s += v * avs[k]; d += v * avd[k]; }
  as_[i] = s; ad_[i] = d;
}

// monotone float <-> uint key so segment-max can use integer atomicMax
__device__ __forceinline__ unsigned f2key(float f) {
  unsigned u = __float_as_uint(f);
  return (u & 0x80000000u) ? ~u : (u | 0x80000000u);
}
__device__ __forceinline__ float key2f(unsigned k) {
  unsigned u = (k & 0x80000000u) ? (k ^ 0x80000000u) : ~k;
  return __uint_as_float(u);
}

// pass 1: e = leaky_relu(as[src]+ad[dst]); store e; segment max over dst
__global__ void k_edge1(const int* __restrict__ src, const int* __restrict__ dst,
                        const float* __restrict__ as_, const float* __restrict__ ad_,
                        float* __restrict__ ebuf, unsigned* __restrict__ mkey,
                        int E, int N) {
  int i = blockIdx.x * blockDim.x + threadIdx.x;
  if (i >= E + N) return;
  int s, d;
  if (i < E) { s = src[i]; d = dst[i]; } else { s = d = i - E; }
  float e = as_[s] + ad_[d];
  e = (e > 0.f) ? e : NEG_SLOPE * e;
  ebuf[i] = e;
  atomicMax(&mkey[d], f2key(e));
}

// pass 2: ex = exp(e - m[dst]); store ex in-place; segment sum over dst
__global__ void k_edge2(const int* __restrict__ dst, const unsigned* __restrict__ mkey,
                        float* __restrict__ ebuf, float* __restrict__ denom,
                        int E, int N) {
  int i = blockIdx.x * blockDim.x + threadIdx.x;
  if (i >= E + N) return;
  int d = (i < E) ? dst[i] : (i - E);
  float ex = __expf(ebuf[i] - key2f(mkey[d]));
  ebuf[i] = ex;
  atomicAdd(&denom[d], ex);
}

// pass 3: out[dst] += (ex/denom[dst]) * h[src]   — H lanes per edge, coalesced
__global__ void k_scatter(const int* __restrict__ src, const int* __restrict__ dst,
                          const float* __restrict__ ebuf, const float* __restrict__ denom,
                          const float* __restrict__ h, float* __restrict__ out,
                          int E, int N, int H) {
  int t = blockIdx.x * blockDim.x + threadIdx.x;
  int i = t / H, j = t % H;
  if (i >= E + N) return;
  int s, d;
  if (i < E) { s = src[i]; d = dst[i]; } else { s = d = i - E; }
  float coef = ebuf[i] / denom[d];
  atomicAdd(&out[(size_t)d * H + j], coef * h[(size_t)s * H + j]);
}

// out1h = f16(relu(out1 + b1))  (layer-2 GEMM operand)
__global__ void k_bias_relu_f16(const float* __restrict__ in, const float* __restrict__ b,
                                _Float16* __restrict__ out, int n, int H) {
  int i = blockIdx.x * blockDim.x + threadIdx.x;
  if (i >= n) return;
  float v = in[i] + b[i % H];
  out[i] = (_Float16)(v > 0.f ? v : 0.f);
}

// y = sigmoid( (relu(out2+b2) + resid + res_b) . fc_w + fc_b )
__global__ void k_final(const float* __restrict__ out2, const float* __restrict__ resid,
                        const float* __restrict__ b2, const float* __restrict__ resb,
                        const float* __restrict__ fcw, const float* __restrict__ fcb,
                        float* __restrict__ y, int n, int H) {
  int i = blockIdx.x * blockDim.x + threadIdx.x;
  if (i >= n) return;
  const float* o = out2  + (size_t)i * H;
  const float* r = resid + (size_t)i * H;
  float acc = fcb[0];
  for (int k = 0; k < H; ++k) {
    float hv = o[k] + b2[k];
    hv = (hv > 0.f ? hv : 0.f) + r[k] + resb[k];
    acc += hv * fcw[k];
  }
  y[i] = 1.0f / (1.0f + __expf(-acc));
}

// ---------------------------------------------------------------- launcher
extern "C" void kernel_launch(void* const* d_in, const int* in_sizes, int n_in,
                              void* d_out, int out_size, void* d_ws, size_t ws_size,
                              hipStream_t stream) {
  const float* x    = (const float*)d_in[0];
  const int*   ei   = (const int*)  d_in[1];
  const float* W1   = (const float*)d_in[2];
  const float* avs1 = (const float*)d_in[3];
  const float* avd1 = (const float*)d_in[4];
  const float* b1   = (const float*)d_in[5];
  const float* W2   = (const float*)d_in[6];
  const float* avs2 = (const float*)d_in[7];
  const float* avd2 = (const float*)d_in[8];
  const float* b2   = (const float*)d_in[9];
  const float* resw = (const float*)d_in[10];
  const float* resb = (const float*)d_in[11];
  const float* fcw  = (const float*)d_in[12];
  const float* fcb  = (const float*)d_in[13];
  float* y = (float*)d_out;

  const int F = 128, H1 = 32, H2 = 64;
  const int N  = in_sizes[0] / F;
  const int E  = in_sizes[1] / 2;
  const int EN = E + N;
  const int* src = ei;
  const int* dst = ei + E;

  // ---- workspace carve-out (lifetime overlays: h2 reuses xh, out1h reuses h1)
  char* ws = (char*)d_ws;
  size_t cur = 0;
  auto alloc = [&](size_t bytes) -> char* {
    cur = (cur + 255) & ~(size_t)255;
    char* p = ws + cur; cur += bytes; return p;
  };
  _Float16* xh   = (_Float16*)alloc((size_t)N * F  * 2);  // dead after the two K=128 GEMMs
  _Float16* W1t  = (_Float16*)alloc((size_t)F  * H1 * 2);
  _Float16* W2t  = (_Float16*)alloc((size_t)H1 * H2 * 2);
  _Float16* Rt   = (_Float16*)alloc((size_t)F  * H2 * 2);
  float*    h1   = (float*)   alloc((size_t)N * H1 * 4);  // dead after layer-1 scatter
  float*    asn  = (float*)   alloc((size_t)N * 4);
  float*    adn  = (float*)   alloc((size_t)N * 4);
  unsigned* mkey = (unsigned*)alloc((size_t)N * 4);
  float*    den  = (float*)   alloc((size_t)N * 4);
  float*    ebuf = (float*)   alloc((size_t)EN * 4);
  float*    out1 = (float*)   alloc((size_t)N * H1 * 4);
  float*    out2 = (float*)   alloc((size_t)N * H2 * 4);
  float*    resid= (float*)   alloc((size_t)N * H2 * 4);
  float*    h2   = (float*)xh;           // N*H2*4 == N*F*2 bytes: exact overlay
  _Float16* out1h= (_Float16*)h1;        // N*H1*2 <= N*H1*4 bytes

  const int T = 256;
  auto cdiv = [](long a, long b) { return (int)((a + b - 1) / b); };

  // ---- f16 conversions
  k_cvt_f16<<<cdiv((long)N * F, T), T, 0, stream>>>(x, xh, N * F);
  k_cvt_transpose_f16<<<cdiv(F * H1, T), T, 0, stream>>>(W1,  W1t, F,  H1);
  k_cvt_transpose_f16<<<cdiv(H1 * H2, T), T, 0, stream>>>(W2, W2t, H1, H2);
  k_cvt_transpose_f16<<<cdiv(F * H2, T), T, 0, stream>>>(resw, Rt, F,  H2);

  // ---- residual GEMM first (so xh region can be reused as h2 later)
  k_wmma_gemm<<<cdiv((long)(N / 16) * (H2 / 16), 4), 128, 0, stream>>>(xh, Rt, resid, N, F, H2);
  // ---- layer-1 linear: h1 = x @ W1
  k_wmma_gemm<<<cdiv((long)(N / 16) * (H1 / 16), 4), 128, 0, stream>>>(xh, W1t, h1, N, F, H1);

  // ---- layer-1 attention + aggregate
  k_alpha<<<cdiv(N, T), T, 0, stream>>>(h1, avs1, avd1, asn, adn, N, H1);
  k_fill_u32<<<cdiv(N, T), T, 0, stream>>>(mkey, 0u, N);
  k_fill_u32<<<cdiv(N, T), T, 0, stream>>>((unsigned*)den, 0u, N);
  k_fill_u32<<<cdiv((long)N * H1, T), T, 0, stream>>>((unsigned*)out1, 0u, N * H1);
  k_edge1<<<cdiv(EN, T), T, 0, stream>>>(src, dst, asn, adn, ebuf, mkey, E, N);
  k_edge2<<<cdiv(EN, T), T, 0, stream>>>(dst, mkey, ebuf, den, E, N);
  k_scatter<<<cdiv((long)EN * H1, T), T, 0, stream>>>(src, dst, ebuf, den, h1, out1, E, N, H1);
  k_bias_relu_f16<<<cdiv((long)N * H1, T), T, 0, stream>>>(out1, b1, out1h, N * H1, H1);

  // ---- layer-2 linear: h2 = relu(out1) @ W2
  k_wmma_gemm<<<cdiv((long)(N / 16) * (H2 / 16), 4), 128, 0, stream>>>(out1h, W2t, h2, N, H1, H2);

  // ---- layer-2 attention + aggregate
  k_alpha<<<cdiv(N, T), T, 0, stream>>>(h2, avs2, avd2, asn, adn, N, H2);
  k_fill_u32<<<cdiv(N, T), T, 0, stream>>>(mkey, 0u, N);
  k_fill_u32<<<cdiv(N, T), T, 0, stream>>>((unsigned*)den, 0u, N);
  k_fill_u32<<<cdiv((long)N * H2, T), T, 0, stream>>>((unsigned*)out2, 0u, N * H2);
  k_edge1<<<cdiv(EN, T), T, 0, stream>>>(src, dst, asn, adn, ebuf, mkey, E, N);
  k_edge2<<<cdiv(EN, T), T, 0, stream>>>(dst, mkey, ebuf, den, E, N);
  k_scatter<<<cdiv((long)EN * H2, T), T, 0, stream>>>(src, dst, ebuf, den, h2, out2, E, N, H2);

  // ---- fused relu + residual + fc + sigmoid
  k_final<<<cdiv(N, T), T, 0, stream>>>(out2, resid, b2, resb, fcw, fcb, y, N, H2);
}
